// GraphUNetCoarsening_14705968021523
// MI455X (gfx1250) — compile-verified, matrix operand order checked by reading, not run on hardware
//
#include <hip/hip_runtime.h>
#include <hip/hip_bf16.h>

// ---------------- problem constants (from reference) ----------------
#define NN   65536      // B * NPG
#define CC   256        // in features
#define HH   512        // hidden
#define EE   524288     // edges
#define BBg  64         // graphs
#define NPGc 1024
#define SPGc 64
#define KSc  3
#define KOc  480        // ceil(0.5*(1024-64))
#define PPc  30912      // BBg*(KSc+KOc)

typedef __attribute__((ext_vector_type(16))) __bf16        v16bf;
typedef __attribute__((ext_vector_type(8)))  float         v8f;
typedef __attribute__((ext_vector_type(4)))  unsigned int  v4u;
typedef __attribute__((ext_vector_type(8)))  int           v8i;
typedef __attribute__((ext_vector_type(4)))  int           v4i;

__device__ __forceinline__ unsigned short f2bf(float f) {
  unsigned int u = __float_as_uint(f);
  unsigned int r = u + 0x7FFFu + ((u >> 16) & 1u);   // round-to-nearest-even
  return (unsigned short)(r >> 16);
}

// ---- TDM: async copy of a 32(k) x 128(row) bf16 tile, row stride CC elems ----
// D# layout per CDNA5 ISA (2D: groups 2/3 zero).
// This toolchain exposes the 6-arg builtin (clang-23 / therock-10.0 headers).
__device__ __forceinline__ void tdm_load_tile32x128(const unsigned short* gsrc,
                                                    unsigned ldsAddr) {
  unsigned long long ga = (unsigned long long)(size_t)gsrc;
  v4u g0;
  g0[0] = 1u;                                            // count=1, user mode
  g0[1] = ldsAddr;                                       // lds_addr
  g0[2] = (unsigned)ga;                                  // global_addr[31:0]
  g0[3] = (unsigned)((ga >> 32) & 0x1FFFFFFu) | (2u << 30); // addr[56:32] | type=2
  v8i g1;
  g1[0] = (int)(1u << 16);                               // wg_mask=0, data_size=1 (2B)
  g1[1] = (int)(32u << 16);                              // tensor_dim0 = 32 (lo16)
  g1[2] = (int)(128u << 16);                             // dim0 hi=0 | tensor_dim1=128
  g1[3] = (int)(32u << 16);                              // dim1 hi=0 | tile_dim0=32
  g1[4] = (int)128u;                                     // tile_dim1=128, tile_dim2=0
  g1[5] = (int)CC;                                       // tensor_dim0_stride = 256 elems
  g1[6] = 0;                                             // stride hi / dim1_stride lo
  g1[7] = 0;
  v4i z4 = {0, 0, 0, 0};
  v8i z8 = {0, 0, 0, 0, 0, 0, 0, 0};
  __builtin_amdgcn_tensor_load_to_lds(g0, g1, z4, z4, z8, 0);
}

// ---------------- degree / norm ----------------
__global__ __launch_bounds__(256) void init_deg_kernel(float* deg) {
  int n = blockIdx.x * 256 + threadIdx.x;
  if (n < NN) deg[n] = 1.0f;                         // self-loop weight
}

__global__ __launch_bounds__(256) void deg_edges_kernel(const int* __restrict__ ei,
                                                        const float* __restrict__ ea,
                                                        float* deg) {
  int e = blockIdx.x * 256 + threadIdx.x;
  if (e < EE) atomicAdd(&deg[ei[EE + e]], ea[e]);    // aggregate at col = dst
}

__global__ __launch_bounds__(256) void dinv_kernel(const float* __restrict__ deg, float* dinv) {
  int n = blockIdx.x * 256 + threadIdx.x;
  if (n < NN) {
    float d = deg[n];
    dinv[n] = d > 0.0f ? rsqrtf(d) : 0.0f;
  }
}

// ---------------- conv1 pre-aggregation: agg = A_norm @ x  (N x C) ----------------
__global__ __launch_bounds__(256) void agg_init_kernel(const float* __restrict__ x,
                                                       const float* __restrict__ dinv,
                                                       float* __restrict__ agg) {
  size_t idx = (size_t)blockIdx.x * 256 + threadIdx.x;        // one float4 each
  size_t n = idx / (CC / 4);
  int    c = (int)(idx % (CC / 4)) * 4;
  float di = dinv[n];
  float sl = di * di;                                          // self-loop norm
  float4 xv = *(const float4*)(x + n * CC + c);
  float4 o = make_float4(sl * xv.x, sl * xv.y, sl * xv.z, sl * xv.w);
  *(float4*)(agg + n * CC + c) = o;
}

__global__ __launch_bounds__(64) void agg_edges_kernel(const float* __restrict__ x,
                                                       const int* __restrict__ ei,
                                                       const float* __restrict__ ea,
                                                       const float* __restrict__ dinv,
                                                       float* __restrict__ agg) {
  int e = blockIdx.x;
  int row = ei[e];
  int col = ei[EE + e];
  float norm = dinv[row] * ea[e] * dinv[col];
  int c = threadIdx.x * 4;
  float4 xv = *(const float4*)(x + (size_t)row * CC + c);
  float* dst = agg + (size_t)col * CC + c;
  atomicAdd(dst + 0, norm * xv.x);
  atomicAdd(dst + 1, norm * xv.y);
  atomicAdd(dst + 2, norm * xv.z);
  atomicAdd(dst + 3, norm * xv.w);
}

// ---------------- one-shot f32 -> bf16 conversions (halve GEMM HBM traffic) ----------------
__global__ __launch_bounds__(256) void to_bf16_kernel(const float* __restrict__ src,
                                                      unsigned short* __restrict__ dst,
                                                      int nquads) {
  int i = blockIdx.x * 256 + threadIdx.x;   // one float4 -> 4 bf16
  if (i < nquads) {
    float4 f = ((const float4*)src)[i];
    unsigned int lo = (unsigned)f2bf(f.x) | ((unsigned)f2bf(f.y) << 16);
    unsigned int hi = (unsigned)f2bf(f.z) | ((unsigned)f2bf(f.w) << 16);
    ((uint2*)dst)[i] = make_uint2(lo, hi);
  }
}

// W1 (CC x HH, row-major) -> W1t bf16 (HH x CC): B tiles become row-strided 2D copies
__global__ __launch_bounds__(256) void w1t_kernel(const float* __restrict__ W1,
                                                  unsigned short* __restrict__ w1t) {
  int t = blockIdx.x * 256 + threadIdx.x;   // t over HH*CC
  int col = t / CC;                          // 0..HH-1
  int k   = t % CC;                          // 0..CC-1
  w1t[t] = f2bf(W1[(size_t)k * HH + col]);
}

// ---------------- WMMA GEMM: h = leakyrelu(aggbf @ W1t^T + b1)  (N x H) ----------------
// 128x128 tile per 256-thread block (8 waves); TDM double-buffered LDS staging.
__global__ __launch_bounds__(256) void gemm1_kernel(const unsigned short* __restrict__ aggbf,
                                                    const unsigned short* __restrict__ w1t,
                                                    const float* __restrict__ b1,
                                                    float* __restrict__ h) {
  __shared__ __align__(16) unsigned short Asm[2][128 * 32];   // [row][k], ping-pong
  __shared__ __align__(16) unsigned short Bsm[2][128 * 32];   // [col][k], ping-pong

  const int tid  = threadIdx.x;
  const int wave = tid >> 5;
  const int lane = tid & 31;
  const int rowBase = blockIdx.x * 128;
  const int colBase = blockIdx.y * 128;

  // LDS byte addresses: generic-pointer low 32 bits are the LDS address (ISA 10.2)
  const unsigned ldsA0 = (unsigned)(size_t)&Asm[0][0];
  const unsigned ldsA1 = (unsigned)(size_t)&Asm[1][0];
  const unsigned ldsB0 = (unsigned)(size_t)&Bsm[0][0];
  const unsigned ldsB1 = (unsigned)(size_t)&Bsm[1][0];

  v8f acc[8] = {};

  const unsigned short* aRow = aggbf + (size_t)rowBase * CC;
  const unsigned short* bRow = w1t  + (size_t)colBase * CC;

  if (wave == 0) {                                   // TDM issues once per wave
    tdm_load_tile32x128(aRow, ldsA0);
    tdm_load_tile32x128(bRow, ldsB0);
  }

  int buf = 0;
  for (int kk = 0; kk < CC; kk += 32) {
    if (wave == 0) {
      if (kk + 32 < CC) {
        tdm_load_tile32x128(aRow + kk + 32, buf ? ldsA0 : ldsA1);
        tdm_load_tile32x128(bRow + kk + 32, buf ? ldsB0 : ldsB1);
        __builtin_amdgcn_s_wait_tensorcnt(2);        // current pair resident (in-order)
      } else {
        __builtin_amdgcn_s_wait_tensorcnt(0);
      }
    }
    __syncthreads();

    // A fragment: lane -> m = lane&15, K runs {k0..k0+7, k0+16..k0+23}, k0 = 8*(lane>>4)
    const int m  = lane & 15;
    const int k0 = (lane >> 4) * 8;
    const unsigned short* As = Asm[buf];
    const unsigned short* Bs = Bsm[buf];
    union Frag { v16bf v; uint4 q[2]; } a;
    a.q[0] = *(const uint4*)&As[(wave * 16 + m) * 32 + k0];
    a.q[1] = *(const uint4*)&As[(wave * 16 + m) * 32 + k0 + 16];

#pragma unroll
    for (int j = 0; j < 8; ++j) {
      union FragB { v16bf v; uint4 q[2]; } b;
      b.q[0] = *(const uint4*)&Bs[(j * 16 + m) * 32 + k0];
      b.q[1] = *(const uint4*)&Bs[(j * 16 + m) * 32 + k0 + 16];
      acc[j] = __builtin_amdgcn_wmma_f32_16x16x32_bf16(
          false, a.v, false, b.v, (short)0, acc[j], false, false);
    }
    __syncthreads();                                 // buffer reuse fence
    buf ^= 1;
  }

  // epilogue: + bias, LeakyReLU, store f32
  const int n     = lane & 15;
  const int mhalf = lane >> 4;
#pragma unroll
  for (int j = 0; j < 8; ++j) {
    int col = colBase + j * 16 + n;
    float bias = b1[col];
#pragma unroll
    for (int v = 0; v < 8; ++v) {
      int row = rowBase + wave * 16 + mhalf * 8 + v;
      float val = acc[j][v] + bias;
      val = val > 0.0f ? val : 0.01f * val;
      h[(size_t)row * HH + col] = val;
    }
  }
}

// ---------------- conv2 matvec: s = h @ W2  (one wave per node) ----------------
__global__ __launch_bounds__(256) void matvec_kernel(const float* __restrict__ h,
                                                     const float* __restrict__ W2,
                                                     float* __restrict__ s) {
  int node = blockIdx.x * 8 + (threadIdx.x >> 5);
  int lane = threadIdx.x & 31;
  const float* hp = h + (size_t)node * HH;
  float sum = 0.0f;
#pragma unroll
  for (int rep = 0; rep < 4; ++rep) {
    int c = lane * 4 + rep * 128;
    float4 hv = *(const float4*)(hp + c);
    float4 wv = *(const float4*)(W2 + c);
    sum += hv.x * wv.x + hv.y * wv.y + hv.z * wv.z + hv.w * wv.w;
  }
#pragma unroll
  for (int off = 16; off > 0; off >>= 1) sum += __shfl_down(sum, off, 32);
  if (lane == 0) s[node] = sum;
}

__global__ __launch_bounds__(256) void score_init_kernel(const float* __restrict__ s,
                                                         const float* __restrict__ dinv,
                                                         const float* __restrict__ b2,
                                                         float* __restrict__ score) {
  int n = blockIdx.x * 256 + threadIdx.x;
  if (n < NN) {
    float di = dinv[n];
    score[n] = di * di * s[n] + b2[0];
  }
}

__global__ __launch_bounds__(256) void score_edges_kernel(const int* __restrict__ ei,
                                                          const float* __restrict__ ea,
                                                          const float* __restrict__ dinv,
                                                          const float* __restrict__ s,
                                                          float* __restrict__ score) {
  int e = blockIdx.x * 256 + threadIdx.x;
  if (e < EE) {
    int row = ei[e];
    int col = ei[EE + e];
    float norm = dinv[row] * ea[e] * dinv[col];
    atomicAdd(&score[col], norm * s[row]);
  }
}

__global__ __launch_bounds__(256) void nanfix_kernel(float* score) {
  int n = blockIdx.x * 256 + threadIdx.x;
  if (n < NN) {
    const float FMAX = 3.4028234663852886e38f;
    float v = score[n];
    if (v != v) v = 0.0f;
    else if (v > FMAX) v = FMAX;
    else if (v < -FMAX) v = -FMAX;
    score[n] = v;
  }
}

// ---------------- exact top-k per graph via rank counting ----------------
__global__ __launch_bounds__(256) void topk_kernel(const float* __restrict__ score,
                                                   int* __restrict__ perm,
                                                   float* __restrict__ out_sent_scores,
                                                   float* __restrict__ out_sent_batch) {
  __shared__ float sg[NPGc];
  const int g = blockIdx.x;
  const int tid = threadIdx.x;
  const float* sc = score + (size_t)g * NPGc;
  for (int i = tid; i < NPGc; i += 256) sg[i] = sc[i];
  __syncthreads();

  for (int i = tid; i < SPGc; i += 256) {
    out_sent_scores[g * SPGc + i] = sg[i];
    out_sent_batch[g * SPGc + i] = (float)g;
  }

  // top-KS of first SPG (ties -> lower index, matching lax.top_k)
  for (int i = tid; i < SPGc; i += 256) {
    float v = sg[i];
    int rank = 0;
    for (int j = 0; j < SPGc; ++j) {
      float u = sg[j];
      rank += (u > v) || (u == v && j < i);
    }
    if (rank < KSc) perm[g * KSc + rank] = g * NPGc + i;
  }

  // top-KO of the remaining NPG-SPG
  for (int i = tid; i < NPGc - SPGc; i += 256) {
    float v = sg[SPGc + i];
    int rank = 0;
    for (int j = 0; j < NPGc - SPGc; ++j) {
      float u = sg[SPGc + j];
      rank += (u > v) || (u == v && j < i);
    }
    if (rank < KOc) perm[BBg * KSc + g * KOc + rank] = g * NPGc + SPGc + i;
  }
}

// ---------------- output assembly ----------------
__global__ __launch_bounds__(256) void mask_init_kernel(int* mask) {
  int n = blockIdx.x * 256 + threadIdx.x;
  if (n < NN) mask[n] = -1;
}

__global__ __launch_bounds__(256) void perm_out_kernel(const int* __restrict__ perm,
                                                       const float* __restrict__ score,
                                                       int* __restrict__ mask,
                                                       float* __restrict__ out_batch,
                                                       float* __restrict__ out_perm,
                                                       float* __restrict__ out_sperm) {
  int k = blockIdx.x * 256 + threadIdx.x;
  if (k < PPc) {
    int node = perm[k];
    mask[node] = k;
    out_perm[k]  = (float)node;
    out_sperm[k] = score[node];
    out_batch[k] = (float)(node >> 10);     // node / NPG
  }
}

__global__ __launch_bounds__(64) void xout_kernel(const float* __restrict__ x,
                                                  const int* __restrict__ perm,
                                                  const float* __restrict__ score,
                                                  float* __restrict__ out_x) {
  int k = blockIdx.x;
  int node = perm[k];
  float sp = score[node];
  int c = threadIdx.x * 4;
  float4 xv = *(const float4*)(x + (size_t)node * CC + c);
  float4 o = make_float4(sp * xv.x, sp * xv.y, sp * xv.z, sp * xv.w);
  *(float4*)(out_x + (size_t)k * CC + c) = o;
}

__global__ __launch_bounds__(256) void edges_out_kernel(const int* __restrict__ ei,
                                                        const float* __restrict__ ea,
                                                        const int* __restrict__ mask,
                                                        float* __restrict__ out_ei,
                                                        float* __restrict__ out_ea,
                                                        float* __restrict__ out_keep) {
  int e = blockIdx.x * 256 + threadIdx.x;
  if (e < EE) {
    int nr = mask[ei[e]];
    int nc = mask[ei[EE + e]];
    bool keep = (nr >= 0) && (nc >= 0);
    out_ei[e]      = keep ? (float)nr : -1.0f;
    out_ei[EE + e] = keep ? (float)nc : -1.0f;
    out_ea[e]      = keep ? ea[e] : 0.0f;
    out_keep[e]    = keep ? 1.0f : 0.0f;
  }
}

// ---------------- launch ----------------
extern "C" void kernel_launch(void* const* d_in, const int* in_sizes, int n_in,
                              void* d_out, int out_size, void* d_ws, size_t ws_size,
                              hipStream_t stream) {
  const float* x  = (const float*)d_in[0];
  const int*   ei = (const int*)d_in[1];
  const float* ea = (const float*)d_in[2];
  const float* W1 = (const float*)d_in[4];
  const float* b1 = (const float*)d_in[5];
  const float* W2 = (const float*)d_in[6];
  const float* b2 = (const float*)d_in[7];
  float* out = (float*)d_out;

  // output section offsets (reference tuple order, flattened)
  constexpr size_t O0 = 0;                          // x_out        P*C
  constexpr size_t O1 = O0 + (size_t)PPc * CC;      // ei_new       2*E
  constexpr size_t O2 = O1 + 2 * (size_t)EE;        // ea_new       E
  constexpr size_t O3 = O2 + (size_t)EE;            // batch_out    P
  constexpr size_t O4 = O3 + (size_t)PPc;           // perm         P
  constexpr size_t O5 = O4 + (size_t)PPc;           // score_perm   P
  constexpr size_t O6 = O5 + (size_t)PPc;           // sent_scores  B*SPG
  constexpr size_t O7 = O6 + (size_t)BBg * SPGc;    // sent_batch   B*SPG
  constexpr size_t O8 = O7 + (size_t)BBg * SPGc;    // keep         E

  // workspace carve
  char* w = (char*)d_ws;
  float* deg   = (float*)w;  w += sizeof(float) * NN;
  float* dinv  = (float*)w;  w += sizeof(float) * NN;
  float* agg   = (float*)w;  w += sizeof(float) * (size_t)NN * CC;
  float* h     = (float*)w;  w += sizeof(float) * (size_t)NN * HH;
  float* s     = (float*)w;  w += sizeof(float) * NN;
  float* score = (float*)w;  w += sizeof(float) * NN;
  int*   mask  = (int*)w;    w += sizeof(int) * NN;
  int*   perm  = (int*)w;    w += sizeof(int) * PPc;
  unsigned short* aggbf = (unsigned short*)w;  w += sizeof(unsigned short) * (size_t)NN * CC;
  unsigned short* w1t   = (unsigned short*)w;  w += sizeof(unsigned short) * (size_t)HH * CC;

  // 1) degrees and symmetric norm
  init_deg_kernel<<<NN / 256, 256, 0, stream>>>(deg);
  deg_edges_kernel<<<EE / 256, 256, 0, stream>>>(ei, ea, deg);
  dinv_kernel<<<NN / 256, 256, 0, stream>>>(deg, dinv);

  // 2) conv1: aggregate-then-GEMM (A x) @ W1, bf16 WMMA + TDM staging
  w1t_kernel<<<(HH * CC) / 256, 256, 0, stream>>>(W1, w1t);
  agg_init_kernel<<<(NN * (CC / 4)) / 256, 256, 0, stream>>>(x, dinv, agg);
  agg_edges_kernel<<<EE, 64, 0, stream>>>(x, ei, ea, dinv, agg);
  to_bf16_kernel<<<(NN * CC / 4) / 256, 256, 0, stream>>>(agg, aggbf, NN * CC / 4);
  gemm1_kernel<<<dim3(NN / 128, HH / 128), 256, 0, stream>>>(aggbf, w1t, b1, h);

  // 3) conv2: matvec-then-aggregate A (h W2) + b2
  matvec_kernel<<<NN / 8, 256, 0, stream>>>(h, W2, s);
  score_init_kernel<<<NN / 256, 256, 0, stream>>>(s, dinv, b2, score);
  score_edges_kernel<<<EE / 256, 256, 0, stream>>>(ei, ea, dinv, s, score);
  nanfix_kernel<<<NN / 256, 256, 0, stream>>>(score);

  // 4) per-graph exact top-k + sentence outputs
  topk_kernel<<<BBg, 256, 0, stream>>>(score, perm, out + O6, out + O7);

  // 5) permutation-indexed outputs
  mask_init_kernel<<<NN / 256, 256, 0, stream>>>(mask);
  perm_out_kernel<<<(PPc + 255) / 256, 256, 0, stream>>>(perm, score, mask,
                                                         out + O3, out + O4, out + O5);
  xout_kernel<<<PPc, 64, 0, stream>>>(x, perm, score, out + O0);
  edges_out_kernel<<<EE / 256, 256, 0, stream>>>(ei, ea, mask,
                                                 out + O1, out + O2, out + O8);
}
